// BC_conv_77137612636466
// MI455X (gfx1250) — 8-bit matrix operand order repaired and checked by reading, compiled, not run on hardware
//
#include <hip/hip_runtime.h>

typedef int v8i __attribute__((ext_vector_type(8)));
typedef int v4b __attribute__((ext_vector_type(4)));
typedef __attribute__((address_space(1))) v4b gv4b;
typedef __attribute__((address_space(3))) v4b lv4b;

constexpr int kH   = 112;
constexpr int kW   = 112;
constexpr int kB   = 16;
constexpr int kC2  = 256;
constexpr int kKT  = 2304;
constexpr int kHW  = kH * kW;
constexpr int kM   = kB * kHW;
constexpr int kKSTEPS = kKT / 64;
constexpr int kStages = kKSTEPS / 2;

__device__ __forceinline__ signed char bsign(float f) {
    float v = f + 1e-6f;
    return (signed char)((v > 0.0f) ? 1 : ((v < 0.0f) ? -1 : 0));
}

__device__ __forceinline__ void async_copy16(void* lds, const void* g) {
#if __has_builtin(__builtin_amdgcn_global_load_async_to_lds_b128)
    __builtin_amdgcn_global_load_async_to_lds_b128(
        (gv4b*)g, (lv4b*)lds, 0, 0);
#else
    *(int4*)lds = *(const int4*)g;
#endif
}

__device__ __forceinline__ void wait_async0() {
#if __has_builtin(__builtin_amdgcn_s_wait_asynccnt)
    __builtin_amdgcn_s_wait_asynccnt(0);
#else
    asm volatile("s_wait_asynccnt 0x0" ::: "memory");
#endif
}

__global__ __launch_bounds__(256) void binarize_x_kernel(const float* __restrict__ x,
                                                         signed char* __restrict__ xb) {
    int tid = blockIdx.x * 256 + threadIdx.x;
    int c4  = tid & 63;
    int bhw = tid >> 6;
    if (bhw >= kM) return;
    int b  = bhw / kHW;
    int hw = bhw - b * kHW;
    int c  = c4 * 4;
    const float* src = x + ((size_t)b * kC2 + c) * kHW + hw;
    char4 v;
    v.x = bsign(src[0 * kHW]);
    v.y = bsign(src[1 * kHW]);
    v.z = bsign(src[2 * kHW]);
    v.w = bsign(src[3 * kHW]);
    ((char4*)xb)[(size_t)bhw * 64 + c4] = v;
}

__global__ __launch_bounds__(256) void binarize_w_kernel(const float* __restrict__ wr,
                                                         const float* __restrict__ wi,
                                                         signed char* __restrict__ wgt) {
    int tid = blockIdx.x * 256 + threadIdx.x;
    int n = tid / kKT;
    int k = tid - n * kKT;
    int khkw = k >> 8;
    int c2   = k & 255;
    int kh = khkw / 3, kw = khkw - kh * 3;
    signed char v;
    if (n < 128) {
        int co = n;
        if (c2 < 128) v =  bsign(wr[(((size_t)co * 128 + c2)        * 3 + kh) * 3 + kw]);
        else          v = (signed char)-bsign(wi[(((size_t)co * 128 + (c2 - 128)) * 3 + kh) * 3 + kw]);
    } else {
        int co = n - 128;
        if (c2 < 128) v =  bsign(wi[(((size_t)co * 128 + c2)        * 3 + kh) * 3 + kw]);
        else          v =  bsign(wr[(((size_t)co * 128 + (c2 - 128)) * 3 + kh) * 3 + kw]);
    }
    wgt[(size_t)n * kKT + k] = v;
}

__global__ __launch_bounds__(256) void bconv_wmma_kernel(const signed char* __restrict__ xb,
                                                         const signed char* __restrict__ wgt,
                                                         const float* __restrict__ bias,
                                                         float* __restrict__ out) {
    __shared__ __align__(16) signed char ldsA[2 * 2 * 64 * 64];
    __shared__ __align__(16) signed char ldsB[2 * 2 * 128 * 64];

    const int tid   = threadIdx.x;
    const int wave  = tid >> 5;
    const int lane  = tid & 31;
    const int pbase = blockIdx.x * 64;
    const int nbase = blockIdx.y * 128;

    const int ar  = tid >> 2;
    const int aq  = tid & 3;
    const int ap  = pbase + ar;
    const int ab  = ap / kHW;
    const int ahw = ap - ab * kHW;
    const int aoh = ahw / kW;
    const int aow = ahw - aoh * kW;

    const int bn = tid >> 1;
    const int bh = tid & 1;
    const signed char* bsrc = wgt + (size_t)(nbase + bn) * kKT + bh * 32;

    const int mi = wave & 3;
    const int ng = wave >> 2;
    const int m0 = mi * 16;
    const int lm    = lane & 15;
    const int lhalf = lane >> 4;

    v8i acc[4] = {};

    auto stage_pair = [&](int it, int buf) {
        const int kk0 = it * 2;
#pragma unroll
        for (int s = 0; s < 2; ++s) {
            const int kk = kk0 + s;
            {
                int khkw = kk >> 2;
                int cblk = kk & 3;
                int kh = khkw / 3, kw = khkw - kh * 3;
                int ih = aoh + kh - 1, iw = aow + kw - 1;
                signed char* dst = ldsA + (buf * 2 + s) * 4096 + ar * 64 + aq * 16;
                if ((unsigned)ih < (unsigned)kH && (unsigned)iw < (unsigned)kW) {
                    const signed char* src =
                        xb + (((size_t)ab * kH + ih) * kW + iw) * kC2 + cblk * 64 + aq * 16;
                    async_copy16(dst, src);
                } else {
                    *(int4*)dst = make_int4(0, 0, 0, 0);
                }
            }
            {
                const signed char* src = bsrc + kk * 64;
                signed char* dst = ldsB + (buf * 2 + s) * 8192 + bn * 64 + bh * 32;
                async_copy16(dst,      src);
                async_copy16(dst + 16, src + 16);
            }
        }
    };

    stage_pair(0, 0);

    for (int it = 0; it < kStages; ++it) {
        const int cur = it & 1;
        wait_async0();
        __syncthreads();
        if (it + 1 < kStages) stage_pair(it + 1, cur ^ 1);

#pragma unroll
        for (int s = 0; s < 2; ++s) {
            const signed char* bufA = ldsA + (cur * 2 + s) * 4096;
            const signed char* bufB = ldsB + (cur * 2 + s) * 8192;

            v8i afrag;
            {
                const signed char* abase = bufA + (m0 + lm) * 64 + lhalf * 16;
                int2 a01 = *(const int2*)(abase + 0);
                int2 a23 = *(const int2*)(abase + 8);
                int2 a45 = *(const int2*)(abase + 32);
                int2 a67 = *(const int2*)(abase + 40);
                afrag[0] = a01.x; afrag[1] = a01.y;
                afrag[2] = a23.x; afrag[3] = a23.y;
                afrag[4] = a45.x; afrag[5] = a45.y;
                afrag[6] = a67.x; afrag[7] = a67.y;
            }
#pragma unroll
            for (int t = 0; t < 4; ++t) {
                const signed char* bbase = bufB + (ng * 64 + t * 16 + lm) * 64 + lhalf * 16;
                int4 b03 = *(const int4*)(bbase);
                int4 b47 = *(const int4*)(bbase + 32);
                v8i bfrag;
                bfrag[0] = b03.x; bfrag[1] = b03.y; bfrag[2] = b03.z; bfrag[3] = b03.w;
                bfrag[4] = b47.x; bfrag[5] = b47.y; bfrag[6] = b47.z; bfrag[7] = b47.w;
                acc[t] = __builtin_amdgcn_wmma_i32_16x16x64_iu8(true, afrag, true, bfrag,
                                                                acc[t], false, false);
            }
        }
    }

    int ob[8], ooh[8], oow[8];
#pragma unroll
    for (int r = 0; r < 8; ++r) {
        int p  = pbase + m0 + lhalf * 8 + r;
        int bb = p / kHW;
        int hw = p - bb * kHW;
        int oh = hw / kW;
        ob[r] = bb; ooh[r] = oh; oow[r] = hw - oh * kW;
    }
#pragma unroll
    for (int t = 0; t < 4; ++t) {
        int n = nbase + ng * 64 + t * 16 + lm;
        float bv = bias[n];
#pragma unroll
        for (int r = 0; r < 8; ++r) {
            size_t oidx = (((size_t)ob[r] * kC2 + n) * kH + ooh[r]) * kW + oow[r];
            out[oidx] = (float)acc[t][r] + bv;
        }
    }
}

extern "C" void kernel_launch(void* const* d_in, const int* in_sizes, int n_in,
                              void* d_out, int out_size, void* d_ws, size_t ws_size,
                              hipStream_t stream) {
    const float* x    = (const float*)d_in[0];
    const float* wr   = (const float*)d_in[1];
    const float* wi   = (const float*)d_in[2];
    const float* bias = (const float*)d_in[3];
    float* out = (float*)d_out;

    signed char* xb  = (signed char*)d_ws;
    signed char* wgt = (signed char*)d_ws + (size_t)kM * kC2;

    binarize_x_kernel<<<(kM * 64) / 256, 256, 0, stream>>>(x, xb);
    binarize_w_kernel<<<(kC2 * kKT) / 256, 256, 0, stream>>>(wr, wi, wgt);
    bconv_wmma_kernel<<<dim3(kM / 64, kC2 / 128), 256, 0, stream>>>(xb, wgt, bias, out);
}
